// PatchClassifier_86895778333415
// MI455X (gfx1250) — matrix operand order repaired and checked by reading, not compiled, not run on hardware
//
#include <hip/hip_runtime.h>
#include <hip/hip_bf16.h>

// ---------------------------------------------------------------------------
// Model constants (PatchClassifier reference)
// ---------------------------------------------------------------------------
#define BV   32000   // vocab
#define DD   512     // model dim
#define HH   8       // heads
#define DHH  64      // head dim
#define PP   16      // patch
#define NLAY 8
#define FFD  2048
#define NCL  4
#define BB   32
#define LL   4096
#define NPAT 256     // L / P
#define SS   257     // NPAT + 1

typedef __attribute__((ext_vector_type(16))) __bf16 v16bf;
typedef __attribute__((ext_vector_type(8)))  float  v8f;

union FragBF {
    uint4 u4[2];
    v16bf v;
};

__device__ __forceinline__ v8f zero8() {
    v8f z;
#pragma unroll
    for (int i = 0; i < 8; ++i) z[i] = 0.0f;
    return z;
}

__device__ __forceinline__ unsigned short f2bf(float f) {
    unsigned int u = __float_as_uint(f);
    unsigned int r = u + 0x7FFFu + ((u >> 16) & 1u);   // round-to-nearest-even
    return (unsigned short)(r >> 16);
}

// Low 32 bits of a generic pointer to LDS == LDS byte offset.
__device__ __forceinline__ unsigned lds_off(const void* p) {
    return (unsigned)(unsigned long long)p;
}

// CDNA5 async global->LDS copy (ASYNCcnt-tracked), 16 bytes per lane.
__device__ __forceinline__ void async_copy_b128(unsigned lds, const void* gptr) {
    asm volatile("global_load_async_to_lds_b128 %0, %1, off"
                 :: "v"(lds), "v"(gptr)
                 : "memory");
}
__device__ __forceinline__ void wait_asynccnt0() {
    asm volatile("s_wait_asynccnt 0x0" ::: "memory");
}

// ---------------------------------------------------------------------------
// bf16 WMMA GEMM, 128(M) x 64(N) block tile, K-step 32, 256 threads (8 waves).
// A: bf16 row-major [M,K] (or embedding gather).  B: bf16 PRE-TRANSPOSED [N,K].
// Staging is pure async-to-LDS b128 (no conversion, no transpose in-kernel).
// ---------------------------------------------------------------------------
enum { EPI_F32 = 0, EPI_RES = 1, EPI_GELU_BF16 = 2, EPI_CONV = 3 };

struct GB {
    const unsigned short* Ab;    // AMODE 0
    const unsigned short* Bt;    // [N][K] bf16
    const unsigned short* embb;  // AMODE 1
    const int*            tok;   // AMODE 1
    const float*          bias;
    const float*          res;   // EPI_RES
    const float*          pos;   // EPI_CONV
    float*                C;
    unsigned short*       Cb;
    int M, K, lda, ldc;
};

template <int AMODE, int EPI>
__global__ __launch_bounds__(256) void gemm_bf16_kernel(GB g) {
    __shared__ __align__(16) unsigned short As[128][32];
    __shared__ __align__(16) unsigned short Bs[64][32];

    const int tid  = threadIdx.x;
    const int m0   = blockIdx.y * 128;
    const int n0   = blockIdx.x * 64;
    const int lane = tid & 31;
    const int wv   = tid >> 5;
    const int wm   = (wv >> 1) * 32;
    const int wn   = (wv & 1) * 32;
    const int half = lane >> 4;
    const int l15  = lane & 15;

    const unsigned asBase = lds_off(&As[0][0]);
    const unsigned bsBase = lds_off(&Bs[0][0]);

    // A tile: 128 rows x 64B; each thread copies 16B at two flat offsets.
    const int fA0 = tid * 16;
    const int fA1 = tid * 16 + 4096;
    const int rA0 = fA0 >> 6, cA0 = (fA0 & 63) >> 1;
    const int rA1 = fA1 >> 6, cA1 = (fA1 & 63) >> 1;
    // B tile: 64 rows x 64B; one 16B copy per thread.
    const int fB = tid * 16;
    const int rB = fB >> 6, cB = (fB & 63) >> 1;

    const int gmA0 = m0 + rA0, gmA1 = m0 + rA1;
    const int gA0c = (gmA0 < g.M) ? gmA0 : g.M - 1;   // clamp: junk rows discarded
    const int gA1c = (gmA1 < g.M) ? gmA1 : g.M - 1;

    const unsigned short* aRow0 = nullptr;
    const unsigned short* aRow1 = nullptr;
    if constexpr (AMODE == 0) {
        aRow0 = g.Ab + (size_t)gA0c * g.lda;
        aRow1 = g.Ab + (size_t)gA1c * g.lda;
    }
    const unsigned short* bRow = g.Bt + (size_t)(n0 + rB) * g.K;

    v8f acc[2][2];
#pragma unroll
    for (int a = 0; a < 2; ++a)
#pragma unroll
        for (int b = 0; b < 2; ++b) acc[a][b] = zero8();

    for (int k0 = 0; k0 < g.K; k0 += 32) {
        const unsigned short *pa0, *pa1;
        if constexpr (AMODE == 0) {
            pa0 = aRow0 + k0 + cA0;
            pa1 = aRow1 + k0 + cA1;
        } else {
            // conv gather: K index kk = p*512 + i; a 32-wide slice stays inside
            // one token's contiguous embedding row.
            const int p  = k0 >> 9;
            const int i0 = k0 & 511;
            const int t0 = g.tok[(gA0c >> 8) * LL + (gA0c & 255) * PP + p];
            const int t1 = g.tok[(gA1c >> 8) * LL + (gA1c & 255) * PP + p];
            pa0 = g.embb + (size_t)t0 * DD + i0 + cA0;
            pa1 = g.embb + (size_t)t1 * DD + i0 + cA1;
        }
        async_copy_b128(asBase + fA0, pa0);
        async_copy_b128(asBase + fA1, pa1);
        async_copy_b128(bsBase + fB, bRow + k0 + cB);
        wait_asynccnt0();
        __syncthreads();

        FragBF af[2], bf[2];
#pragma unroll
        for (int mt = 0; mt < 2; ++mt) {
            const int row = wm + mt * 16 + l15;
            af[mt].u4[0] = *(const uint4*)&As[row][half * 8];
            af[mt].u4[1] = *(const uint4*)&As[row][16 + half * 8];
        }
#pragma unroll
        for (int nt = 0; nt < 2; ++nt) {
            const int col = wn + nt * 16 + l15;
            bf[nt].u4[0] = *(const uint4*)&Bs[col][half * 8];
            bf[nt].u4[1] = *(const uint4*)&Bs[col][16 + half * 8];
        }
#pragma unroll
        for (int mt = 0; mt < 2; ++mt)
#pragma unroll
            for (int nt = 0; nt < 2; ++nt)
                acc[mt][nt] = __builtin_amdgcn_wmma_f32_16x16x32_bf16(
                    false, af[mt].v, false, bf[nt].v,
                    (short)0, acc[mt][nt], false, false);
        __syncthreads();
    }

#pragma unroll
    for (int mt = 0; mt < 2; ++mt) {
#pragma unroll
        for (int nt = 0; nt < 2; ++nt) {
            const int gn = n0 + wn + nt * 16 + l15;
#pragma unroll
            for (int r = 0; r < 8; ++r) {
                const int gm = m0 + wm + mt * 16 + r + half * 8;
                if (gm >= g.M) continue;
                float v = acc[mt][nt][r] + g.bias[gn];
                if constexpr (EPI == EPI_F32) {
                    g.C[(size_t)gm * g.ldc + gn] = v;
                } else if constexpr (EPI == EPI_RES) {
                    g.C[(size_t)gm * g.ldc + gn] = v + g.res[(size_t)gm * g.ldc + gn];
                } else if constexpr (EPI == EPI_GELU_BF16) {
                    v = 0.5f * v * (1.0f + erff(v * 0.70710678118654752f));
                    g.Cb[(size_t)gm * g.ldc + gn] = f2bf(v);
                } else {  // EPI_CONV: row (b,nn) -> x[b, nn+1, :] + pos
                    const int b = gm >> 8, nn = gm & 255;
                    v += g.pos[(size_t)(nn + 1) * DD + gn];
                    const size_t off = ((size_t)b * SS + nn + 1) * DD + gn;
                    g.C[off]  = v;
                    g.Cb[off] = f2bf(v);
                }
            }
        }
    }
}

// ---------------------------------------------------------------------------
// kv[b,h,d,m] = sum_s kf[b,s,h,d] * v[b,s,h,m]   (64x64 per (b,h), WMMA)
// ---------------------------------------------------------------------------
__global__ __launch_bounds__(128) void kv_wmma_kernel(const float* __restrict__ kf,
                                                      const float* __restrict__ vv,
                                                      float* __restrict__ kv) {
    const int bh = blockIdx.x;
    const int b  = bh >> 3, h = bh & 7;
    const float* kbase = kf + ((size_t)b * SS * HH + h) * DHH;
    const float* vbase = vv + ((size_t)b * SS * HH + h) * DHH;

    __shared__ __align__(16) unsigned short As[64][32];  // [d][s]
    __shared__ __align__(16) unsigned short Bs[64][32];  // [m][s]

    const int tid = threadIdx.x;
    const int lane = tid & 31, wv = tid >> 5;
    const int half = lane >> 4, l15 = lane & 15;

    v8f acc[4];
#pragma unroll
    for (int i = 0; i < 4; ++i) acc[i] = zero8();

    for (int s0 = 0; s0 < SS; s0 += 32) {
        const int d  = tid >> 1;
        const int ss = (tid & 1) * 16;
#pragma unroll 4
        for (int j = 0; j < 16; ++j) {
            const int s = s0 + ss + j;
            float av = 0.0f, bvv = 0.0f;
            if (s < SS) {
                av  = kbase[(size_t)s * DD + d];
                bvv = vbase[(size_t)s * DD + d];
            }
            As[d][ss + j] = f2bf(av);
            Bs[d][ss + j] = f2bf(bvv);
        }
        __syncthreads();

        FragBF af;
        const int arow = wv * 16 + l15;
        af.u4[0] = *(const uint4*)&As[arow][half * 8];
        af.u4[1] = *(const uint4*)&As[arow][16 + half * 8];
#pragma unroll
        for (int nt = 0; nt < 4; ++nt) {
            FragBF bf;
            const int col = nt * 16 + l15;
            bf.u4[0] = *(const uint4*)&Bs[col][half * 8];
            bf.u4[1] = *(const uint4*)&Bs[col][16 + half * 8];
            acc[nt] = __builtin_amdgcn_wmma_f32_16x16x32_bf16(
                false, af.v, false, bf.v, (short)0, acc[nt], false, false);
        }
        __syncthreads();
    }

    float* out = kv + (size_t)bh * DHH * DHH;
#pragma unroll
    for (int nt = 0; nt < 4; ++nt)
#pragma unroll
        for (int r = 0; r < 8; ++r)
            out[(size_t)(wv * 16 + r + half * 8) * DHH + nt * 16 + l15] = acc[nt][r];
}

// ---------------------------------------------------------------------------
// attn[b,s,h,m] = z[b,s,h] * sum_d qf[b,s,h,d] * kv[b,h,d,m]  -> bf16 output
// ---------------------------------------------------------------------------
__global__ __launch_bounds__(128) void attn_wmma_kernel(const float* __restrict__ qf,
                                                        const float* __restrict__ kv,
                                                        const float* __restrict__ z,
                                                        unsigned short* __restrict__ attnb) {
    const int MB = (SS + 63) / 64;  // 5
    const int blk = blockIdx.x;
    const int mb = blk % MB;
    const int bh = blk / MB;
    const int b = bh >> 3, h = bh & 7;
    const int m0 = mb * 64;

    const float* qbase = qf + ((size_t)b * SS * HH + h) * DHH;
    const float* kvp   = kv + (size_t)bh * DHH * DHH;

    __shared__ __align__(16) unsigned short As[64][32];
    __shared__ __align__(16) unsigned short Bs[64][32];

    const int tid = threadIdx.x;
    const int lane = tid & 31, wv = tid >> 5;
    const int half = lane >> 4, l15 = lane & 15;

    v8f acc[4];
#pragma unroll
    for (int i = 0; i < 4; ++i) acc[i] = zero8();

#pragma unroll
    for (int k0 = 0; k0 < DHH; k0 += 32) {
        const int r  = tid >> 1;
        const int ks = (tid & 1) * 16;
        const int s  = m0 + r;
#pragma unroll 4
        for (int j = 0; j < 16; ++j) {
            const int kk = k0 + ks + j;
            As[r][ks + j] = f2bf((s < SS) ? qbase[(size_t)s * DD + kk] : 0.0f);
            Bs[r][ks + j] = f2bf(kvp[(size_t)kk * DHH + r]);
        }
        __syncthreads();

        FragBF af;
        const int arow = wv * 16 + l15;
        af.u4[0] = *(const uint4*)&As[arow][half * 8];
        af.u4[1] = *(const uint4*)&As[arow][16 + half * 8];
#pragma unroll
        for (int nt = 0; nt < 4; ++nt) {
            FragBF bf;
            const int col = nt * 16 + l15;
            bf.u4[0] = *(const uint4*)&Bs[col][half * 8];
            bf.u4[1] = *(const uint4*)&Bs[col][16 + half * 8];
            acc[nt] = __builtin_amdgcn_wmma_f32_16x16x32_bf16(
                false, af.v, false, bf.v, (short)0, acc[nt], false, false);
        }
        __syncthreads();
    }

#pragma unroll
    for (int nt = 0; nt < 4; ++nt)
#pragma unroll
        for (int r = 0; r < 8; ++r) {
            const int s = m0 + wv * 16 + r + half * 8;
            if (s < SS) {
                const float zz = z[((size_t)b * SS + s) * HH + h];
                attnb[(((size_t)b * SS + s) * HH + h) * DHH + nt * 16 + l15] =
                    f2bf(acc[nt][r] * zz);
            }
        }
}

// ---------------------------------------------------------------------------
// Conversion kernels (run once per launch; trivial vs GEMM FLOPs)
// ---------------------------------------------------------------------------
__global__ void cvt_plain_kernel(const float* __restrict__ in,
                                 unsigned short* __restrict__ out, long total) {
    const long idx = (long)blockIdx.x * blockDim.x + threadIdx.x;
    if (idx < total) out[idx] = f2bf(in[idx]);
}

// W: [L][K][N] f32  ->  out: [L][N][K] bf16
__global__ void cvt_transpose_kernel(const float* __restrict__ W,
                                     unsigned short* __restrict__ out,
                                     int Lb, int K, int N) {
    const long idx = (long)blockIdx.x * blockDim.x + threadIdx.x;
    const long total = (long)Lb * K * N;
    if (idx >= total) return;
    const int l = (int)(idx / ((long)K * N));
    const int rem = (int)(idx % ((long)K * N));
    const int n = rem / K;
    const int k = rem % K;
    out[idx] = f2bf(W[(size_t)l * K * N + (size_t)k * N + n]);
}

// conv_w [o][i][p] f32 -> out [o][p*512+i] bf16 (i.e., [N=512][K=8192])
__global__ void cvt_convw_kernel(const float* __restrict__ cw,
                                 unsigned short* __restrict__ out) {
    const int idx = blockIdx.x * blockDim.x + threadIdx.x;  // 512*8192
    if (idx >= DD * PP * DD) return;
    const int o = idx >> 13;
    const int r = idx & 8191;
    const int p = r >> 9;
    const int i = r & 511;
    out[idx] = f2bf(cw[((size_t)o * DD + i) * PP + p]);
}

// ---------------------------------------------------------------------------
// Small support kernels
// ---------------------------------------------------------------------------
__global__ void lengths_kernel(const float* __restrict__ mask, float* __restrict__ len) {
    __shared__ float red[256];
    const int b = blockIdx.x, tid = threadIdx.x;
    float s = 0.0f;
    for (int i = tid; i < LL; i += 256) s += mask[(size_t)b * LL + i];
    red[tid] = s;
    __syncthreads();
    for (int off = 128; off > 0; off >>= 1) {
        if (tid < off) red[tid] += red[tid + off];
        __syncthreads();
    }
    if (tid == 0) len[b] = ceilf((red[0] + 1.0f) / (float)PP);
}

__global__ void cls_pos_kernel(const float* __restrict__ cls,
                               const float* __restrict__ pos,
                               float* __restrict__ x,
                               unsigned short* __restrict__ xb) {
    const int idx = blockIdx.x * blockDim.x + threadIdx.x;   // B*D
    if (idx >= BB * DD) return;
    const int b = idx >> 9, d = idx & 511;
    const float v = cls[d] + pos[d];
    const size_t off = ((size_t)b * SS) * DD + d;
    x[off]  = v;
    xb[off] = f2bf(v);
}

__global__ void rope_elu_kernel(float* __restrict__ q, float* __restrict__ k,
                                const float* __restrict__ len) {
    int t = blockIdx.x * blockDim.x + threadIdx.x;  // B*S*H*32 exactly
    const int j = t & 31;  t >>= 5;
    const int h = t & 7;   t >>= 3;
    const int s = t % SS;
    const int b = t / SS;
    if (b >= BB) return;

    const size_t base = (((size_t)b * SS + s) * HH + h) * DHH;
    const float inv = __expf(-(float)j * 0.28782313662425572f);  // ln(1e4)/32
    const float fr  = (float)s * inv;
    const float c = __cosf(fr), sn = __sinf(fr);

    const float q1 = q[base + j], q2 = q[base + j + 32];
    const float k1 = k[base + j], k2 = k[base + j + 32];
    float qa = q1 * c - q2 * sn, qb = q2 * c + q1 * sn;
    float ka = k1 * c - k2 * sn, kb = k2 * c + k1 * sn;
    qa = (qa > 0.0f) ? qa + 1.0f : __expf(qa);
    qb = (qb > 0.0f) ? qb + 1.0f : __expf(qb);
    ka = (ka > 0.0f) ? ka + 1.0f : __expf(ka);
    kb = (kb > 0.0f) ? kb + 1.0f : __expf(kb);
    const float lm = ((float)s < len[b]) ? 1.0f : 0.0f;
    q[base + j] = qa;        q[base + j + 32] = qb;
    k[base + j] = ka * lm;   k[base + j + 32] = kb * lm;
}

__global__ void ksum_kernel(const float* __restrict__ kf, float* __restrict__ ks) {
    const int bh = blockIdx.x, d = threadIdx.x;  // grid B*H, block 64
    const int b = bh >> 3, h = bh & 7;
    const float* kb = kf + ((size_t)b * SS * HH + h) * DHH + d;
    float s = 0.0f;
    for (int i = 0; i < SS; ++i) s += kb[(size_t)i * DD];
    ks[(size_t)bh * DHH + d] = s;
}

__global__ void z_kernel(const float* __restrict__ qf, const float* __restrict__ ks,
                         float* __restrict__ z) {
    const int idx = blockIdx.x * blockDim.x + threadIdx.x;  // B*S*H exactly
    if (idx >= BB * SS * HH) return;
    const int h = idx % HH;
    const int s = (idx / HH) % SS;
    const int b = idx / (SS * HH);
    const float* qb = qf + (((size_t)b * SS + s) * HH + h) * DHH;
    const float* kb = ks + ((size_t)(b * HH + h)) * DHH;
    float acc = 0.0f;
#pragma unroll 8
    for (int d = 0; d < DHH; ++d) acc += qb[d] * kb[d];
    z[idx] = 1.0f / (acc + 1e-6f);
}

// LayerNorm, optional bf16 secondary output.
__global__ __launch_bounds__(256) void layernorm_kernel(const float* __restrict__ in,
                                                        const float* __restrict__ sc,
                                                        const float* __restrict__ bt,
                                                        float* __restrict__ out,
                                                        unsigned short* __restrict__ outb) {
    __shared__ float red[256];
    const int row = blockIdx.x, tid = threadIdx.x;
    const size_t base = (size_t)row * DD;
    const float e0 = in[base + tid], e1 = in[base + tid + 256];
    red[tid] = e0 + e1;
    __syncthreads();
    for (int off = 128; off > 0; off >>= 1) {
        if (tid < off) red[tid] += red[tid + off];
        __syncthreads();
    }
    const float mean = red[0] * (1.0f / (float)DD);
    __syncthreads();
    const float d0 = e0 - mean, d1 = e1 - mean;
    red[tid] = d0 * d0 + d1 * d1;
    __syncthreads();
    for (int off = 128; off > 0; off >>= 1) {
        if (tid < off) red[tid] += red[tid + off];
        __syncthreads();
    }
    const float rs = rsqrtf(red[0] * (1.0f / (float)DD) + 1e-5f);
    const float o0 = d0 * rs * sc[tid]       + bt[tid];
    const float o1 = d1 * rs * sc[tid + 256] + bt[tid + 256];
    out[base + tid]       = o0;
    out[base + tid + 256] = o1;
    if (outb) {
        outb[base + tid]       = f2bf(o0);
        outb[base + tid + 256] = f2bf(o1);
    }
}

__global__ void head_kernel(const float* __restrict__ x,
                            const float* __restrict__ w,
                            const float* __restrict__ bia,
                            float* __restrict__ out) {
    const int idx = threadIdx.x;  // 128 = B*NC
    if (idx >= BB * NCL) return;
    const int b = idx >> 2, c = idx & 3;
    const float* xr = x + ((size_t)b * SS) * DD;  // row s=0
    float acc = 0.0f;
#pragma unroll 8
    for (int d = 0; d < DD; ++d) acc += xr[d] * w[(size_t)d * NCL + c];
    out[idx] = acc + bia[c];
}

// ---------------------------------------------------------------------------
// Host-side orchestration
// ---------------------------------------------------------------------------
template <int AMODE, int EPI>
static inline void launch_gemm(const GB& g, int N, hipStream_t stream) {
    dim3 grid(N / 64, (g.M + 127) / 128);
    gemm_bf16_kernel<AMODE, EPI><<<grid, 256, 0, stream>>>(g);
}

extern "C" void kernel_launch(void* const* d_in, const int* in_sizes, int n_in,
                              void* d_out, int out_size, void* d_ws, size_t ws_size,
                              hipStream_t stream) {
    const int*   inputs  = (const int*)  d_in[0];
    const float* imask   = (const float*)d_in[1];
    const float* emb     = (const float*)d_in[2];
    const float* conv_w  = (const float*)d_in[3];
    const float* conv_b  = (const float*)d_in[4];
    const float* pos     = (const float*)d_in[5];
    const float* cls     = (const float*)d_in[6];
    const float* Wq      = (const float*)d_in[7];
    const float* bq      = (const float*)d_in[8];
    const float* Wk      = (const float*)d_in[9];
    const float* bk      = (const float*)d_in[10];
    const float* Wv      = (const float*)d_in[11];
    const float* bv      = (const float*)d_in[12];
    const float* Wo      = (const float*)d_in[13];
    const float* bo      = (const float*)d_in[14];
    const float* ln1_s   = (const float*)d_in[15];
    const float* ln1_b   = (const float*)d_in[16];
    const float* ln2_s   = (const float*)d_in[17];
    const float* ln2_b   = (const float*)d_in[18];
    const float* W1      = (const float*)d_in[19];
    const float* b1      = (const float*)d_in[20];
    const float* W2      = (const float*)d_in[21];
    const float* b2      = (const float*)d_in[22];
    const float* lnf_s   = (const float*)d_in[23];
    const float* lnf_b   = (const float*)d_in[24];
    const float* out_w   = (const float*)d_in[25];
    const float* out_b   = (const float*)d_in[26];
    (void)in_sizes; (void)n_in; (void)out_size; (void)ws_size;

    const size_t NTOK = (size_t)BB * SS;          // 8224
    const size_t XSZ  = NTOK * DD;                // 4,210,688

    char* base = (char*)d_ws;
    size_t off = 0;
    auto alloc_f = [&](size_t n) {
        float* p = (float*)(base + off);
        off += (n * 4 + 255) & ~(size_t)255;
        return p;
    };
    auto alloc_h = [&](size_t n) {
        unsigned short* p = (unsigned short*)(base + off);
        off += (n * 2 + 255) & ~(size_t)255;
        return p;
    };

    // f32 activations
    float* X   = alloc_f(XSZ);
    float* Qf  = alloc_f(XSZ);
    float* Kf  = alloc_f(XSZ);
    float* Vf  = alloc_f(XSZ);   // also reused as pre-LN buffer
    float* KV  = alloc_f((size_t)BB * HH * DHH * DHH);
    float* KS  = alloc_f((size_t)BB * HH * DHH);
    float* ZZ  = alloc_f((size_t)BB * SS * HH);
    float* LEN = alloc_f(BB);
    // bf16 operands
    unsigned short* Xb   = alloc_h(XSZ);
    unsigned short* ATTb = alloc_h(XSZ);
    unsigned short* Hb   = alloc_h(NTOK * FFD);
    unsigned short* Eb   = alloc_h((size_t)BV * DD);
    unsigned short* Ct   = alloc_h((size_t)DD * PP * DD);
    unsigned short* Wqt  = alloc_h((size_t)NLAY * DD * DD);
    unsigned short* Wkt  = alloc_h((size_t)NLAY * DD * DD);
    unsigned short* Wvt  = alloc_h((size_t)NLAY * DD * DD);
    unsigned short* Wot  = alloc_h((size_t)NLAY * DD * DD);
    unsigned short* W1t  = alloc_h((size_t)NLAY * DD * FFD);
    unsigned short* W2t  = alloc_h((size_t)NLAY * FFD * DD);

    // ---- operand conversion ----
    {
        const long etot = (long)BV * DD;
        cvt_plain_kernel<<<(unsigned)((etot + 255) / 256), 256, 0, stream>>>(emb, Eb, etot);
        cvt_convw_kernel<<<(DD * PP * DD + 255) / 256, 256, 0, stream>>>(conv_w, Ct);
        const long wtot = (long)NLAY * DD * DD;
        const unsigned wg = (unsigned)((wtot + 255) / 256);
        cvt_transpose_kernel<<<wg, 256, 0, stream>>>(Wq, Wqt, NLAY, DD, DD);
        cvt_transpose_kernel<<<wg, 256, 0, stream>>>(Wk, Wkt, NLAY, DD, DD);
        cvt_transpose_kernel<<<wg, 256, 0, stream>>>(Wv, Wvt, NLAY, DD, DD);
        cvt_transpose_kernel<<<wg, 256, 0, stream>>>(Wo, Wot, NLAY, DD, DD);
        const long ftot = (long)NLAY * DD * FFD;
        const unsigned fg = (unsigned)((ftot + 255) / 256);
        cvt_transpose_kernel<<<fg, 256, 0, stream>>>(W1, W1t, NLAY, DD, FFD);
        cvt_transpose_kernel<<<fg, 256, 0, stream>>>(W2, W2t, NLAY, FFD, DD);
    }

    lengths_kernel<<<BB, 256, 0, stream>>>(imask, LEN);
    cls_pos_kernel<<<(BB * DD + 255) / 256, 256, 0, stream>>>(cls, pos, X, Xb);

    // ---- patch conv: (B*n, 8192) x (8192, 512), A gathered from bf16 emb ----
    {
        GB g = {};
        g.embb = Eb; g.tok = inputs; g.Bt = Ct;
        g.bias = conv_b; g.pos = pos; g.C = X; g.Cb = Xb;
        g.M = BB * NPAT; g.K = PP * DD; g.lda = 0; g.ldc = DD;
        launch_gemm<1, EPI_CONV>(g, DD, stream);
    }

    const int MTOK = (int)NTOK;
    for (int i = 0; i < NLAY; ++i) {
        const size_t wOff = (size_t)i * DD * DD;
        const size_t bOff = (size_t)i * DD;

        GB g = {};
        g.Ab = Xb; g.M = MTOK; g.K = DD; g.lda = DD; g.ldc = DD;
        g.Bt = Wqt + wOff; g.bias = bq + bOff; g.C = Qf;
        launch_gemm<0, EPI_F32>(g, DD, stream);
        g.Bt = Wkt + wOff; g.bias = bk + bOff; g.C = Kf;
        launch_gemm<0, EPI_F32>(g, DD, stream);
        g.Bt = Wvt + wOff; g.bias = bv + bOff; g.C = Vf;
        launch_gemm<0, EPI_F32>(g, DD, stream);

        rope_elu_kernel<<<(BB * SS * HH * 32) / 256, 256, 0, stream>>>(Qf, Kf, LEN);
        ksum_kernel<<<BB * HH, 64, 0, stream>>>(Kf, KS);
        kv_wmma_kernel<<<BB * HH, 128, 0, stream>>>(Kf, Vf, KV);
        z_kernel<<<(BB * SS * HH + 255) / 256, 256, 0, stream>>>(Qf, KS, ZZ);
        attn_wmma_kernel<<<BB * HH * ((SS + 63) / 64), 128, 0, stream>>>(Qf, KV, ZZ, ATTb);

        // x = x + attn @ Wo + bo   (Vf reused as pre-LN buffer from here on)
        GB go = {};
        go.Ab = ATTb; go.Bt = Wot + wOff; go.bias = bo + bOff; go.res = X;
        go.C = Vf; go.M = MTOK; go.K = DD; go.lda = DD; go.ldc = DD;
        launch_gemm<0, EPI_RES>(go, DD, stream);
        layernorm_kernel<<<MTOK, 256, 0, stream>>>(Vf, ln1_s + bOff, ln1_b + bOff, X, Xb);

        // h = gelu(x @ W1 + b1) -> bf16
        GB g1 = {};
        g1.Ab = Xb; g1.Bt = W1t + (size_t)i * DD * FFD; g1.bias = b1 + (size_t)i * FFD;
        g1.Cb = Hb; g1.M = MTOK; g1.K = DD; g1.lda = DD; g1.ldc = FFD;
        launch_gemm<0, EPI_GELU_BF16>(g1, FFD, stream);

        // y = x + h @ W2 + b2
        GB g2 = {};
        g2.Ab = Hb; g2.Bt = W2t + (size_t)i * FFD * DD; g2.bias = b2 + bOff;
        g2.res = X; g2.C = Vf; g2.M = MTOK; g2.K = FFD; g2.lda = FFD; g2.ldc = DD;
        launch_gemm<0, EPI_RES>(g2, DD, stream);
        layernorm_kernel<<<MTOK, 256, 0, stream>>>(Vf, ln2_s + bOff, ln2_b + bOff, X, Xb);
    }

    // final LN + head
    layernorm_kernel<<<MTOK, 256, 0, stream>>>(X, lnf_s, lnf_b, Qf, nullptr);
    head_kernel<<<1, 128, 0, stream>>>(Qf, out_w, out_b, (float*)d_out);
}
